// GATBaseline_61194694033411
// MI455X (gfx1250) — matrix operand order repaired and checked by reading, not compiled, not run on hardware
//
#include <hip/hip_runtime.h>
#include <hip/hip_bf16.h>
#include <stdint.h>

typedef __attribute__((ext_vector_type(16))) _Float16 v16h;
typedef __attribute__((ext_vector_type(8)))  _Float16 v8h;
typedef __attribute__((ext_vector_type(8)))  float    v8f;

#define BATCH 64
#define NNODE 400
#define DIM   128
#define NROWS (BATCH * NNODE)      // 25600
#define ADJW  13                   // ceil(400/32) u32 words per (b,t)
#define EPS   1e-5f

__device__ __forceinline__ v16h join8(v8h lo, v8h hi) {
    return __builtin_shufflevector(lo, hi, 0,1,2,3,4,5,6,7,8,9,10,11,12,13,14,15);
}

__device__ __forceinline__ void async_cp16(unsigned lds_addr, const void* gaddr) {
    asm volatile("global_load_async_to_lds_b128 %0, %1, off" :: "v"(lds_addr), "v"(gaddr) : "memory");
}
// Wait until this wave's ASYNCcnt <= n (completions are in order -> drains older).
__device__ __forceinline__ void wait_async(int n) {
    if (n == 0)      asm volatile("s_wait_asynccnt 0" ::: "memory");
    else if (n == 2) asm volatile("s_wait_asynccnt 2" ::: "memory");
    else if (n == 3) asm volatile("s_wait_asynccnt 3" ::: "memory");
    else             asm volatile("s_wait_asynccnt 4" ::: "memory");
}

__device__ __forceinline__ v16h wmma16(v16h a, v16h b, v8f& acc) {
    acc = __builtin_amdgcn_wmma_f32_16x16x32_f16(false, a, false, b, (short)0, acc, false, false);
    return a;
}

// ---------------------------------------------------------------------------
// One-time per call: W[K,N] (f32) -> WT[N,K] (f16), LDS tile transpose
// ---------------------------------------------------------------------------
__global__ void w_to_f16T(const float* __restrict__ W, _Float16* __restrict__ WT, int K, int N) {
    __shared__ _Float16 tile[32][33];
    int kt = blockIdx.x * 32, nt = blockIdx.y * 32;
    int tx = threadIdx.x & 31, ty = threadIdx.x >> 5;   // 256 thr: ty = 0..7
    #pragma unroll
    for (int j = 0; j < 32; j += 8) {
        int k = kt + ty + j, n = nt + tx;
        tile[ty + j][tx] = (k < K && n < N) ? (_Float16)W[(size_t)k * N + n] : (_Float16)0.f;
    }
    __syncthreads();
    #pragma unroll
    for (int j = 0; j < 32; j += 8) {
        int n = nt + ty + j, k = kt + tx;
        if (n < N && k < K) WT[(size_t)n * K + k] = tile[tx][ty + j];
    }
}

// ---------------------------------------------------------------------------
// Adjacency bitmask: bit i of adj[(b*400+t)*13 + w]  <=>  edge (s = w*32+i) -> t
// ---------------------------------------------------------------------------
__global__ void adjmask_kernel(const float* __restrict__ sc, unsigned int* __restrict__ adj) {
    int idx = blockIdx.x * blockDim.x + threadIdx.x;
    if (idx >= NROWS * ADJW) return;
    int w  = idx % ADJW;
    int bt = idx / ADJW;
    int t  = bt % NNODE, b = bt / NNODE;
    unsigned int bits = 0u;
    #pragma unroll 4
    for (int i = 0; i < 32; ++i) {
        int s = w * 32 + i;
        if (s < NNODE) {
            float v = sc[((size_t)b * NNODE + s) * NNODE + t];
            if (v != 0.f || s == t) bits |= (1u << i);
        }
    }
    adj[idx] = bits;
}

// ---------------------------------------------------------------------------
// Feature GEMM: H[25600,128](f16) = A[25600,K] x W[K,128], B from WT[128,K] f16.
// 32-row M-tiles: grid 800, block 256 (8 waves; wave w -> cols [16w,16w+16),
// 2 WMMAs/stage sharing the B fragment). Double-buffered async pipeline.
// ---------------------------------------------------------------------------
template <typename TA>
__global__ void gemm_h_kernel(const TA* __restrict__ A, const _Float16* __restrict__ WT,
                              _Float16* __restrict__ Hout, int K, int lda) {
    __shared__ __align__(16) _Float16 Ash[2][32][40];     // [m][k]
    __shared__ __align__(16) _Float16 BshT[2][DIM][40];   // [n][k]
    const int tid  = threadIdx.x;
    const int wave = tid >> 5, lane = tid & 31;
    const int m0   = blockIdx.x * 32;
    const int n0   = wave * 16;
    const int m    = lane & 15, hh = lane >> 4;
    const int kTiles = (K + 31) / 32;

    auto stage = [&](int kt, int bi) -> int {
        const int k0 = kt * 32;
        if (k0 + 32 <= K) {                              // uniform fast path
            #pragma unroll
            for (int j = 0; j < 2; ++j) {                // B: 512 16B chunks, 2/thread
                int idx = tid + j * 256;
                int c = idx >> 2, koff = (idx & 3) * 8;
                async_cp16((unsigned)(uintptr_t)&BshT[bi][c][koff],
                           &WT[(size_t)c * K + k0 + koff]);
            }
            if constexpr (__is_same(TA, _Float16)) {
                // A: 128 16B chunks; lanes 0..15 of EVERY wave issue one
                int chunk = wave * 16 + lane;            // valid for lane < 16
                int r = (chunk >> 2) & 31, coff = (chunk & 3) * 8;
                const _Float16* gp = &A[(size_t)(m0 + r) * lda + k0 + coff];
                unsigned la = (unsigned)(uintptr_t)&Ash[bi][r][coff];
                if (lane < 16) async_cp16(la, gp);
                return 3;
            } else {
                int idx = tid * 4;                       // A: 1024 elems, 4/thread
                int r = idx >> 5, c = idx & 31;
                const TA* ap = &A[(size_t)(m0 + r) * lda + k0 + c];
                #pragma unroll
                for (int i = 0; i < 4; ++i) Ash[bi][r][c + i] = (_Float16)ap[i];
                return 2;
            }
        } else {                                         // K tail: branchless clamp+select
            {   // A: 4 elems/thread
                int idx = tid * 4;
                int r = idx >> 5, c = idx & 31;
                #pragma unroll
                for (int i = 0; i < 4; ++i) {
                    int kk = k0 + c + i;
                    _Float16 v = (_Float16)A[(size_t)(m0 + r) * lda + min(kk, K - 1)];
                    Ash[bi][r][c + i] = (kk < K) ? v : (_Float16)0.f;
                }
            }
            v8h hz = {};
            #pragma unroll
            for (int j = 0; j < 2; ++j) {                // B: 16B chunks, clamped
                int idx = tid + j * 256;
                int c = idx >> 2, koff = (idx & 3) * 8;
                v8h bv = *(const v8h*)&WT[(size_t)c * K + min(k0 + koff, K - 8)];
                if (k0 + koff + 8 > K) bv = hz;
                *(v8h*)&BshT[bi][c][koff] = bv;
            }
            return 0;
        }
    };

    v8f acc0 = {}, acc1 = {};
    stage(0, 0);
    for (int kt = 0; kt < kTiles; ++kt) {
        int pend_next = 0;
        if (kt + 1 < kTiles) pend_next = stage(kt + 1, (kt + 1) & 1);
        wait_async(pend_next);
        __syncthreads();
        const int bi = kt & 1;
        v16h bm = join8(*(const v8h*)&BshT[bi][n0 + m][hh * 8],
                        *(const v8h*)&BshT[bi][n0 + m][16 + hh * 8]);
        v16h a0 = join8(*(const v8h*)&Ash[bi][m][hh * 8],
                        *(const v8h*)&Ash[bi][m][16 + hh * 8]);
        v16h a1 = join8(*(const v8h*)&Ash[bi][16 + m][hh * 8],
                        *(const v8h*)&Ash[bi][16 + m][16 + hh * 8]);
        wmma16(a0, bm, acc0);
        wmma16(a1, bm, acc1);
        __syncthreads();
    }
    #pragma unroll
    for (int r = 0; r < 8; ++r) {
        Hout[(size_t)(m0 + r + 8 * hh) * DIM + n0 + m]      = (_Float16)acc0[r];
        Hout[(size_t)(m0 + 16 + r + 8 * hh) * DIM + n0 + m] = (_Float16)acc1[r];
    }
}

// ---------------------------------------------------------------------------
// es/ed per node (grid 25600, blk 128); HH = heads (4 or 1), C = 128/HH
// ---------------------------------------------------------------------------
template <int HH>
__global__ void es_ed_kernel(const _Float16* __restrict__ Hf, const float* __restrict__ a_src,
                             const float* __restrict__ a_dst, float* __restrict__ es,
                             float* __restrict__ ed) {
    constexpr int C = DIM / HH;
    __shared__ float s1[128], s2[128];
    int row = blockIdx.x, t = threadIdx.x;
    float hv = (float)Hf[(size_t)row * DIM + t];
    s1[t] = hv * a_src[t];
    s2[t] = hv * a_dst[t];
    __syncthreads();
    #pragma unroll
    for (int s = C >> 1; s > 0; s >>= 1) {
        if ((t & (C - 1)) < s) { s1[t] += s1[t + s]; s2[t] += s2[t + s]; }
        __syncthreads();
    }
    if ((t & (C - 1)) == 0) {
        int head = t / C;
        es[(size_t)row * HH + head] = s1[t];
        ed[(size_t)row * HH + head] = s2[t];
    }
}

// ---------------------------------------------------------------------------
// Softmax stats over sources s for each (b,t,head): running max and sum(exp)
// ---------------------------------------------------------------------------
template <int HH>
__global__ void softmax_stats_kernel(const float* __restrict__ es, const float* __restrict__ ed,
                                     const unsigned int* __restrict__ adj,
                                     float* __restrict__ rowmax, float* __restrict__ rowsum) {
    int row  = blockIdx.x;                 // b*400 + t
    int b    = row / NNODE;
    int wave = threadIdx.x >> 5, lane = threadIdx.x & 31;
    if (wave >= HH) return;
    float edv = ed[(size_t)row * HH + wave];
    const unsigned int* aw = adj + (size_t)row * ADJW;
    float mx = -1e30f;
    for (int s = lane; s < NNODE; s += 32) {
        if ((aw[s >> 5] >> (s & 31)) & 1u) {
            float l = es[((size_t)b * NNODE + s) * HH + wave] + edv;
            l = (l >= 0.f) ? l : 0.2f * l;
            mx = fmaxf(mx, l);
        }
    }
    #pragma unroll
    for (int off = 16; off; off >>= 1) mx = fmaxf(mx, __shfl_xor(mx, off, 32));
    float sum = 0.f;
    for (int s = lane; s < NNODE; s += 32) {
        if ((aw[s >> 5] >> (s & 31)) & 1u) {
            float l = es[((size_t)b * NNODE + s) * HH + wave] + edv;
            l = (l >= 0.f) ? l : 0.2f * l;
            sum += __expf(l - mx);
        }
    }
    #pragma unroll
    for (int off = 16; off; off >>= 1) sum += __shfl_xor(sum, off, 32);
    if (lane == 0) {
        rowmax[(size_t)row * HH + wave] = mx;
        rowsum[(size_t)row * HH + wave] = sum;
    }
}

// ---------------------------------------------------------------------------
// Attention aggregation (WMMA) + bias + BatchNorm + optional ELU.
// grid (25 t-tiles, 64 batches), block 128 = 4 waves; wave w owns c-tiles 2w,2w+1
// (same head), so the exp-rebuilt A fragment feeds 2 WMMAs. Async double buffer.
// ---------------------------------------------------------------------------
template <int HH, bool ELU>
__global__ void attn_aggregate_kernel(const _Float16* __restrict__ Hf, const float* __restrict__ es,
                                      const float* __restrict__ ed, const float* __restrict__ rowmax,
                                      const float* __restrict__ rowsum, const unsigned int* __restrict__ adj,
                                      const float* __restrict__ bias,
                                      const float* __restrict__ bn_g, const float* __restrict__ bn_b,
                                      const float* __restrict__ bn_m, const float* __restrict__ bn_v,
                                      _Float16* __restrict__ Xout) {
    __shared__ __align__(16) _Float16 hsh[2][32][136];   // h[s-chunk][c] (row 272B)
    __shared__ float es_sh[2][32][4];
    const int b    = blockIdx.y;
    const int t0   = blockIdx.x * 16;
    const int tid  = threadIdx.x, wave = tid >> 5, lane = tid & 31;
    const int n    = lane & 15, hh = lane >> 4;
    const int c0a  = wave * 32;                          // first of 2 c-tiles
    const int head = (HH == 4) ? wave : 0;
    const size_t trow = (size_t)(b * NNODE + t0 + n);
    const float ed_t   = ed[trow * HH + head];
    const float mx     = rowmax[trow * HH + head];
    const float sm     = rowsum[trow * HH + head];
    const float inv_sm = (sm > 0.f) ? 1.f / sm : 0.f;

    auto stage = [&](int kc, int bi) -> int {
        const int s0 = kc * 32;
        int ret;
        if (s0 + 32 <= NNODE) {
            #pragma unroll
            for (int j = 0; j < 4; ++j) {                // 512 16B chunks, 4/thread
                int idx = tid + j * 128;
                int r = idx >> 4, coff = (idx & 15) * 8;
                async_cp16((unsigned)(uintptr_t)&hsh[bi][r][coff],
                           &Hf[((size_t)(b * NNODE + s0 + r)) * DIM + coff]);
            }
            ret = 4;
        } else {                                         // tail: branchless clamp+select
            v8h hz = {};
            #pragma unroll
            for (int j = 0; j < 4; ++j) {
                int idx = tid + j * 128;
                int r = idx >> 4, coff = (idx & 15) * 8;
                int rc = min(s0 + r, NNODE - 1) - s0;    // clamp to last valid row
                v8h hv = *(const v8h*)&Hf[((size_t)(b * NNODE + s0 + rc)) * DIM + coff];
                if (s0 + r >= NNODE) hv = hz;
                *(v8h*)&hsh[bi][r][coff] = hv;
            }
            ret = 0;
        }
        if constexpr (HH == 4) {                         // es slab: 32 x 4
            int r = tid >> 2, h2 = tid & 3, s = s0 + r;
            float v = es[((size_t)(b * NNODE + min(s, NNODE - 1))) * HH + h2];
            es_sh[bi][r][h2] = (s < NNODE) ? v : 0.f;
        } else {
            if (tid < 32) {
                int s = s0 + tid;
                float v = es[(size_t)(b * NNODE + min(s, NNODE - 1))];
                es_sh[bi][tid][0] = (s < NNODE) ? v : 0.f;
            }
        }
        return ret;
    };

    v8f accv[2] = {{}, {}};
    stage(0, 0);
    for (int kc = 0; kc < ADJW; ++kc) {
        int pend_next = 0;
        if (kc + 1 < ADJW) pend_next = stage(kc + 1, (kc + 1) & 1);
        wait_async(pend_next);
        __syncthreads();
        const int bi = kc & 1;
        const unsigned int aw = adj[trow * ADJW + kc];
        v16h a;
        #pragma unroll
        for (int e = 0; e < 8; ++e) {
            int i1 = hh * 8 + e;            // K slot 0..7 / 8..15
            int i2 = 16 + hh * 8 + e;       // K slot 16..23 / 24..31
            float a1 = 0.f, a2 = 0.f;
            if ((aw >> i1) & 1u) {
                float l = es_sh[bi][i1][head] + ed_t;
                l = (l >= 0.f) ? l : 0.2f * l;
                a1 = __expf(l - mx) * inv_sm;
            }
            if ((aw >> i2) & 1u) {
                float l = es_sh[bi][i2][head] + ed_t;
                l = (l >= 0.f) ? l : 0.2f * l;
                a2 = __expf(l - mx) * inv_sm;
            }
            a[e]     = (_Float16)a1;
            a[e + 8] = (_Float16)a2;
        }
        #pragma unroll
        for (int ct = 0; ct < 2; ++ct) {
            v16h bm;
            #pragma unroll
            for (int e = 0; e < 16; ++e) bm[e] = hsh[bi][hh * 8 + e + (e & 8)][c0a + ct * 16 + n];
            wmma16(a, bm, accv[ct]);
        }
        __syncthreads();
    }
    #pragma unroll
    for (int ct = 0; ct < 2; ++ct) {
        const int ch = c0a + ct * 16 + n;
        const float scale = __frsqrt_rn(bn_v[ch] + EPS) * bn_g[ch];
        #pragma unroll
        for (int r = 0; r < 8; ++r) {
            int trow_o = t0 + r + 8 * hh;
            float v = accv[ct][r] + bias[ch];
            v = (v - bn_m[ch]) * scale + bn_b[ch];
            if constexpr (ELU) v = (v > 0.f) ? v : (__expf(v) - 1.f);
            Xout[((size_t)(b * NNODE + trow_o)) * DIM + ch] = (_Float16)v;
        }
    }
}

// ---------------------------------------------------------------------------
// Classifier WMMA GEMM, split-K with f32 atomics: C += A[M,K] x BT[N,K]^T (+bias on z==0)
// grid (M/16, N/16, splits), block 32. kPerSplit multiple of 32; C pre-zeroed.
// ---------------------------------------------------------------------------
template <typename TA>
__global__ void gemm_wmma_splitk(const TA* __restrict__ A, const _Float16* __restrict__ BT,
                                 const float* __restrict__ bias, float* __restrict__ Cm,
                                 int K, int lda, int ldc, int kPerSplit) {
    const int lane = threadIdx.x & 31;
    const int m0 = blockIdx.x * 16, n0 = blockIdx.y * 16;
    const int m = lane & 15, hh = lane >> 4;
    const int kBeg = blockIdx.z * kPerSplit;
    const int kEnd = min(K, kBeg + kPerSplit);
    v8f acc = {};
    for (int k0 = kBeg; k0 < kEnd; k0 += 32) {
        if (k0 + 32 < kEnd) __builtin_prefetch(&BT[(size_t)(n0 + m) * K + k0 + 32], 0, 0);
        v16h a, bm;
        if constexpr (__is_same(TA, _Float16)) {
            const _Float16* Arow = A + (size_t)(m0 + m) * lda + k0;
            a = join8(*(const v8h*)&Arow[hh * 8], *(const v8h*)&Arow[16 + hh * 8]);
        } else {
            const TA* Arow = A + (size_t)(m0 + m) * lda + k0;
            #pragma unroll
            for (int e = 0; e < 8; ++e) {
                a[e]     = (_Float16)Arow[hh * 8 + e];
                a[e + 8] = (_Float16)Arow[16 + hh * 8 + e];
            }
        }
        const _Float16* Brow = BT + (size_t)(n0 + m) * K + k0;
        bm = join8(*(const v8h*)&Brow[hh * 8], *(const v8h*)&Brow[16 + hh * 8]);
        wmma16(a, bm, acc);
    }
    #pragma unroll
    for (int r = 0; r < 8; ++r) {
        int row = m0 + r + 8 * hh, col = n0 + m;
        float v = acc[r] + ((blockIdx.z == 0 && bias) ? bias[col] : 0.f);
        atomicAdd(&Cm[(size_t)row * ldc + col], v);
    }
}

// ---------------------------------------------------------------------------
// LayerNorm + ReLU over rows of z[64, F]; grid = 64, block = F (F <= 256, pow2)
// ---------------------------------------------------------------------------
__global__ void ln_relu_kernel(float* __restrict__ z, const float* __restrict__ g,
                               const float* __restrict__ be, int F) {
    __shared__ float sh[256];
    int row = blockIdx.x, t = threadIdx.x;
    float v = z[(size_t)row * F + t];
    sh[t] = v; __syncthreads();
    for (int s = F >> 1; s; s >>= 1) { if (t < s) sh[t] += sh[t + s]; __syncthreads(); }
    float mu = sh[0] / (float)F; __syncthreads();
    float d = v - mu;
    sh[t] = d * d; __syncthreads();
    for (int s = F >> 1; s; s >>= 1) { if (t < s) sh[t] += sh[t + s]; __syncthreads(); }
    float var = sh[0] / (float)F;
    float y = d * __frsqrt_rn(var + EPS) * g[t] + be[t];
    z[(size_t)row * F + t] = fmaxf(y, 0.f);
}

// ---------------------------------------------------------------------------
// Final tiny linear: out[64,2] = z2[64,64] x W3[64,2] + b3  (1 block, 128 thr)
// ---------------------------------------------------------------------------
__global__ void final_linear_kernel(const float* __restrict__ z2, const float* __restrict__ W3,
                                    const float* __restrict__ b3, float* __restrict__ out) {
    int idx = threadIdx.x;               // 128 threads: (b, j)
    int b = idx >> 1, j = idx & 1;
    float acc = b3[j];
    #pragma unroll 8
    for (int k = 0; k < 64; ++k) acc += z2[b * 64 + k] * W3[k * 2 + j];
    out[b * 2 + j] = acc;
}

// ---------------------------------------------------------------------------
extern "C" void kernel_launch(void* const* d_in, const int* in_sizes, int n_in,
                              void* d_out, int out_size, void* d_ws, size_t ws_size,
                              hipStream_t stream) {
    (void)in_sizes; (void)n_in; (void)out_size; (void)ws_size;
    const float* sc = (const float*)d_in[1];
    const float* convW[3], *convAs[3], *convAd[3], *convBias[3];
    const float* bnG[3], *bnB[3], *bnM[3], *bnV[3];
    for (int j = 0; j < 3; ++j) {
        int base = 2 + j * 4;
        convW[j]    = (const float*)d_in[base + 0];
        convAs[j]   = (const float*)d_in[base + 1];
        convAd[j]   = (const float*)d_in[base + 2];
        convBias[j] = (const float*)d_in[base + 3];
        int bnb = 14 + j * 4;
        bnG[j] = (const float*)d_in[bnb + 0];
        bnB[j] = (const float*)d_in[bnb + 1];
        bnM[j] = (const float*)d_in[bnb + 2];
        bnV[j] = (const float*)d_in[bnb + 3];
    }
    const float* W1  = (const float*)d_in[26];
    const float* b1  = (const float*)d_in[27];
    const float* g1  = (const float*)d_in[28];
    const float* be1 = (const float*)d_in[29];
    const float* W2  = (const float*)d_in[30];
    const float* b2  = (const float*)d_in[31];
    const float* g2  = (const float*)d_in[32];
    const float* be2 = (const float*)d_in[33];
    const float* W3  = (const float*)d_in[34];
    const float* b3  = (const float*)d_in[35];

    char* ws = (char*)d_ws;
    size_t off = 0;
    auto alloc = [&](size_t bytes) { void* p = ws + off; off = (off + bytes + 255) & ~(size_t)255; return p; };
    _Float16*     hbuf   = (_Float16*)alloc((size_t)NROWS * DIM * 2);
    _Float16*     xbuf   = (_Float16*)alloc((size_t)NROWS * DIM * 2);
    float*        es     = (float*)alloc((size_t)NROWS * 4 * 4);
    float*        ed     = (float*)alloc((size_t)NROWS * 4 * 4);
    float*        rowmax = (float*)alloc((size_t)NROWS * 4 * 4);
    float*        rowsum = (float*)alloc((size_t)NROWS * 4 * 4);
    unsigned int* adj    = (unsigned int*)alloc((size_t)NROWS * ADJW * 4);
    float*        z1     = (float*)alloc((size_t)BATCH * 256 * 4);
    float*        z2     = (float*)alloc((size_t)BATCH * 64 * 4);
    _Float16*     WTl    = (_Float16*)alloc((size_t)DIM * NNODE * 2);       // per-layer W^T f16
    _Float16*     WTc1   = (_Float16*)alloc((size_t)256 * NNODE * DIM * 2); // W1^T f16 (26MB)
    _Float16*     WTc2   = (_Float16*)alloc((size_t)64 * 256 * 2);          // W2^T f16

    adjmask_kernel<<<(NROWS * ADJW + 255) / 256, 256, 0, stream>>>(sc, adj);

    for (int j = 0; j < 3; ++j) {
        const int Kin = (j == 0) ? NNODE : DIM;
        w_to_f16T<<<dim3((Kin + 31) / 32, (DIM + 31) / 32), 256, 0, stream>>>(convW[j], WTl, Kin, DIM);
        if (j == 0)
            gemm_h_kernel<float><<<NROWS / 32, 256, 0, stream>>>(sc, WTl, hbuf, NNODE, NNODE);
        else
            gemm_h_kernel<_Float16><<<NROWS / 32, 256, 0, stream>>>(xbuf, WTl, hbuf, DIM, DIM);
        if (j < 2) {
            es_ed_kernel<4><<<NROWS, 128, 0, stream>>>(hbuf, convAs[j], convAd[j], es, ed);
            softmax_stats_kernel<4><<<NROWS, 128, 0, stream>>>(es, ed, adj, rowmax, rowsum);
            attn_aggregate_kernel<4, true><<<dim3(NNODE / 16, BATCH), 128, 0, stream>>>(
                hbuf, es, ed, rowmax, rowsum, adj, convBias[j],
                bnG[j], bnB[j], bnM[j], bnV[j], xbuf);
        } else {
            es_ed_kernel<1><<<NROWS, 128, 0, stream>>>(hbuf, convAs[j], convAd[j], es, ed);
            softmax_stats_kernel<1><<<NROWS, 128, 0, stream>>>(es, ed, adj, rowmax, rowsum);
            attn_aggregate_kernel<1, false><<<dim3(NNODE / 16, BATCH), 128, 0, stream>>>(
                hbuf, es, ed, rowmax, rowsum, adj, convBias[j],
                bnG[j], bnB[j], bnM[j], bnV[j], xbuf);
        }
    }

    // Classifier head
    const int K1 = NNODE * DIM;   // 51200
    w_to_f16T<<<dim3((K1 + 31) / 32, (256 + 31) / 32), 256, 0, stream>>>(W1, WTc1, K1, 256);
    hipMemsetAsync(z1, 0, (size_t)BATCH * 256 * 4, stream);
    gemm_wmma_splitk<_Float16><<<dim3(BATCH / 16, 256 / 16, 16), 32, 0, stream>>>(
        xbuf, WTc1, b1, z1, K1, K1, 256, K1 / 16);
    ln_relu_kernel<<<BATCH, 256, 0, stream>>>(z1, g1, be1, 256);

    w_to_f16T<<<dim3((256 + 31) / 32, (64 + 31) / 32), 256, 0, stream>>>(W2, WTc2, 256, 64);
    hipMemsetAsync(z2, 0, (size_t)BATCH * 64 * 4, stream);
    gemm_wmma_splitk<float><<<dim3(BATCH / 16, 64 / 16, 1), 32, 0, stream>>>(
        z1, WTc2, b2, z2, 256, 256, 64, 256);
    ln_relu_kernel<<<BATCH, 64, 0, stream>>>(z2, g2, be2, 64);
    final_linear_kernel<<<1, 128, 0, stream>>>(z2, W3, b3, (float*)d_out);
}
